// UserInteractionPredictor_71854802862397
// MI455X (gfx1250) — compile-verified, matrix operand order checked by reading, not run on hardware
//
#include <hip/hip_runtime.h>
#include <hip/hip_bf16.h>
#include <math.h>
#include <stdint.h>

typedef __attribute__((ext_vector_type(16))) _Float16 v16h;
typedef __attribute__((ext_vector_type(8)))  float    v8f;
typedef unsigned __attribute__((ext_vector_type(4))) uint32x4;
typedef unsigned __attribute__((ext_vector_type(8))) uint32x8;

#define S_SNAP 5
#define NUSR   50000
#define NEVT   10000
#define NEDGE  800000
#define FDIM   64
#define HDIM   128
#define DDIM   128
#define NHEAD  4
#define DHEAD  32
#define NLT    2
#define DFF    512
#define BSZ    8192
#define TT     4
#define EPSV   1e-5f

// rows padded to tile height (128) so GEMM A-tiles can be async-copied
// without bounds checks (over-read rows only feed discarded outputs)
#define NUP    50048    // ceil(50000/128)*128
#define NEP    10112    // ceil(10000/128)*128
#define MSEQ   (TT * NUSR)      // 200000
#define MSEQP  200064           // ceil(200000/128)*128

// ---------------------------------------------------------------------------
// WMMA GEMM: C[M,N] = A[M,K] @ B[K,N] (+C if accum) (+bias[n] if bias!=null)
//   A : f32 [M,K] row-major, buffer padded to >= ceil(M/128)*128 rows
//   Bt: f32 [N,K] (B pre-transposed), N % 64 == 0, K % 32 == 0
// Double-buffered staging:
//   A tile : GLOBAL_LOAD_ASYNC_TO_LDS_B128 (ASYNCcnt, 4 ops/thread/tile)
//   B tile : TENSOR_LOAD_TO_LDS (TDM, wave 0 only; pad_interval=32dw,
//            pad_amount=4dw matches the LDS row stride), TENSORcnt
// f32->f16 in registers, f32 accumulate via v_wmma_f32_16x16x32_f16.
// Block 256 (8 waves); tile 128(M) x 64(N), BK=32; wave w owns rows [16w,16w+16).
// ---------------------------------------------------------------------------
#define LDAF 36   // LDS row stride in floats (32 + 4 pad) = 144B, 16B-aligned

__global__ __launch_bounds__(256)
void gemm_f32_wmma(const float* __restrict__ A, const float* __restrict__ Bt,
                   float* __restrict__ C, const float* __restrict__ bias,
                   int M, int N, int K, int accum)
{
    __shared__ alignas(16) float ldsA[2][128 * LDAF];
    __shared__ alignas(16) float ldsB[2][64 * LDAF];
    const int tid  = threadIdx.x;
    const int lane = tid & 31;
    const int wave = tid >> 5;
    const int m0   = blockIdx.x * 128;
    const int n0   = blockIdx.y * 64;

    v8f acc[4];
#pragma unroll
    for (int t = 0; t < 4; ++t) acc[t] = (v8f)0.f;

    const int mloc = wave * 16 + (lane & 15);
    const int kb   = (lane & 16) ? 8 : 0;   // K sub-offset for lanes 16-31
    const int rA   = tid >> 3;              // 0..31: staging row
    const int cA   = (tid & 7) * 4;         // 0,4,...,28: float offset in row

    // ---- A tile: per-lane async global->LDS copies (ASYNCcnt) ------------
    auto stageA = [&](int buf, int kt) {
#pragma unroll
        for (int it = 0; it < 4; ++it) {
            int row = rA + it * 32;
            const float* g = A + (size_t)(m0 + row) * K + kt + cA;
            unsigned lo = (unsigned)(uintptr_t)&ldsA[buf][row * LDAF + cA];
            asm volatile("global_load_async_to_lds_b128 %0, %1, off"
                         :: "v"(lo), "v"(g) : "memory");
        }
    };

    // ---- B tile: Tensor Data Mover, wave 0 issues one 2-D descriptor -----
    auto stageB = [&](int buf, int kt) {
        if (wave == 0) {
            uint64_t ga = (uint64_t)(uintptr_t)(Bt + (size_t)n0 * K + kt);
            unsigned ldsaddr = (unsigned)(uintptr_t)&ldsB[buf][0];
            unsigned dim0 = (unsigned)(K - kt);   // remaining K (no OOB)
            unsigned dim1 = (unsigned)(N - n0);   // remaining rows (no OOB)
            uint32x4 g0;
            g0[0] = (unsigned)__builtin_amdgcn_readfirstlane(1u);  // count=1
            g0[1] = (unsigned)__builtin_amdgcn_readfirstlane(ldsaddr);
            g0[2] = (unsigned)__builtin_amdgcn_readfirstlane((unsigned)ga);
            g0[3] = (unsigned)__builtin_amdgcn_readfirstlane(
                        (unsigned)((ga >> 32) & 0x1FFFFFFu) | (2u << 30)); // type=2
            uint32x8 g1;
            // data_size=4B(2), pad_enable, pad_interval=32dw(4), pad_amount=4dw(3)
            g1[0] = (unsigned)__builtin_amdgcn_readfirstlane(
                        (2u << 16) | (1u << 20) | (4u << 22) | (3u << 25));
            g1[1] = (unsigned)__builtin_amdgcn_readfirstlane((dim0 & 0xFFFFu) << 16);
            g1[2] = (unsigned)__builtin_amdgcn_readfirstlane(
                        (dim0 >> 16) | ((dim1 & 0xFFFFu) << 16));
            g1[3] = (unsigned)__builtin_amdgcn_readfirstlane(
                        (dim1 >> 16) | (32u << 16));          // tile_dim0=32
            g1[4] = (unsigned)__builtin_amdgcn_readfirstlane(64u); // tile_dim1=64
            g1[5] = (unsigned)__builtin_amdgcn_readfirstlane((unsigned)K); // stride0
            g1[6] = (unsigned)__builtin_amdgcn_readfirstlane(0u);
            g1[7] = (unsigned)__builtin_amdgcn_readfirstlane(0u);
            asm volatile("tensor_load_to_lds %0, %1" :: "s"(g0), "s"(g1) : "memory");
        }
    };

    // ---- 16x16 WMMA compute on one staged K-slice -------------------------
    auto compute = [&](int buf) {
        const float* pa = &ldsA[buf][mloc * LDAF + kb];
        float4 a0 = *(const float4*)(pa + 0);
        float4 a1 = *(const float4*)(pa + 4);
        float4 a2 = *(const float4*)(pa + 16);
        float4 a3 = *(const float4*)(pa + 20);
        float av[16] = {a0.x,a0.y,a0.z,a0.w, a1.x,a1.y,a1.z,a1.w,
                        a2.x,a2.y,a2.z,a2.w, a3.x,a3.y,a3.z,a3.w};
        v16h af;
#pragma unroll
        for (int i = 0; i < 16; ++i) af[i] = (_Float16)av[i];
#pragma unroll
        for (int nt = 0; nt < 4; ++nt) {
            const float* pb = &ldsB[buf][(nt * 16 + (lane & 15)) * LDAF + kb];
            float4 b0 = *(const float4*)(pb + 0);
            float4 b1 = *(const float4*)(pb + 4);
            float4 b2 = *(const float4*)(pb + 16);
            float4 b3 = *(const float4*)(pb + 20);
            float bv[16] = {b0.x,b0.y,b0.z,b0.w, b1.x,b1.y,b1.z,b1.w,
                            b2.x,b2.y,b2.z,b2.w, b3.x,b3.y,b3.z,b3.w};
            v16h bf;
#pragma unroll
            for (int i = 0; i < 16; ++i) bf[i] = (_Float16)bv[i];
            acc[nt] = __builtin_amdgcn_wmma_f32_16x16x32_f16(
                false, af, false, bf, (short)0, acc[nt], false, false);
        }
    };

    // ---- double-buffered pipeline ----------------------------------------
    const int KT = K >> 5;
    stageA(0, 0);
    stageB(0, 0);
    for (int t = 0; t < KT; ++t) {
        int cur  = t & 1;
        int more = (t + 1 < KT);
        if (more) {
            stageA(cur ^ 1, (t + 1) << 5);
            stageB(cur ^ 1, (t + 1) << 5);
            __builtin_prefetch(A + (size_t)(m0 + rA) * K + ((t + 1) << 5), 0, 1);
            asm volatile("s_wait_asynccnt 4" ::: "memory");   // prev A-stage done
        } else {
            asm volatile("s_wait_asynccnt 0" ::: "memory");
        }
        if (wave == 0) {
            if (more) __builtin_amdgcn_s_wait_tensorcnt(1);   // prev B-DMA done
            else      __builtin_amdgcn_s_wait_tensorcnt(0);
        }
        __syncthreads();
        compute(cur);
        __syncthreads();
    }

    // ---- epilogue: C/D layout: VGPR r -> M=r (lanes 0-15) / 8+r (16-31) ---
    const int mrow = m0 + wave * 16 + ((lane & 16) ? 8 : 0);
#pragma unroll
    for (int nt = 0; nt < 4; ++nt) {
        int n = n0 + nt * 16 + (lane & 15);
        float bvs = bias ? bias[n] : 0.f;
#pragma unroll
        for (int r = 0; r < 8; ++r) {
            int m = mrow + r;
            if (m < M) {
                size_t o = (size_t)m * N + n;
                float v = acc[nt][r] + bvs;
                if (accum) v += C[o];
                C[o] = v;
            }
        }
    }
}

// ---------------------------------------------------------------------------
// Weight pre-transpose: Bt[n*K+k] = B[k*N+n] (tiny matrices, run once)
// ---------------------------------------------------------------------------
__global__ void transpose_f32(const float* __restrict__ B, float* __restrict__ Bt,
                              int K, int N)
{
    int gid = blockIdx.x * 256 + threadIdx.x;
    if (gid >= K * N) return;
    int k = gid / N, n = gid - k * N;
    Bt[(size_t)n * K + k] = B[gid];
}

// ---------------------------------------------------------------------------
// SAGE mean-aggregation: agg[dst[e]] += xsrc[src[e]], cnt[dst[e]] += 1
// ---------------------------------------------------------------------------
__global__ __launch_bounds__(256)
void scatter_add(const float* __restrict__ xsrc, const int* __restrict__ src,
                 const int* __restrict__ dst, float* __restrict__ agg,
                 float* __restrict__ cnt, int E)
{
    int gid = blockIdx.x * 256 + threadIdx.x;
    int e = gid >> 5;
    if (e >= E) return;
    int c  = (gid & 31) * 4;
    int si = src[e], di = dst[e];
    const float4 v = *(const float4*)(xsrc + (size_t)si * HDIM + c);
    float* o = agg + (size_t)di * HDIM + c;
    atomicAdd(o + 0, v.x); atomicAdd(o + 1, v.y);
    atomicAdd(o + 2, v.z); atomicAdd(o + 3, v.w);
    if (c == 0) atomicAdd(&cnt[di], 1.0f);
}

__global__ void mean_div(float* __restrict__ agg, const float* __restrict__ cnt, int total)
{
    int gid = blockIdx.x * 256 + threadIdx.x;
    if (gid >= total) return;
    float c = cnt[gid >> 7];
    agg[gid] /= (c < 1.f ? 1.f : c);
}

// ---------------------------------------------------------------------------
// BatchNorm (training batch stats, biased var) + ReLU, two passes.
// ---------------------------------------------------------------------------
__global__ __launch_bounds__(256)
void bn_stats(const float* __restrict__ x, float* __restrict__ stats, int M)
{
    __shared__ float ssum[256], ssq[256];
    int tid = threadIdx.x;
    int c = tid & 127;
    float s = 0.f, q = 0.f;
    for (int r = blockIdx.x * 2 + (tid >> 7); r < M; r += gridDim.x * 2) {
        float v = x[(size_t)r * HDIM + c];
        s += v; q += v * v;
    }
    ssum[tid] = s; ssq[tid] = q;
    __syncthreads();
    if (tid < 128) {
        atomicAdd(&stats[c],       ssum[tid] + ssum[tid + 128]);
        atomicAdd(&stats[128 + c], ssq[tid] + ssq[tid + 128]);
    }
}

__global__ void bn_apply_relu(const float* __restrict__ x, float* __restrict__ y,
                              const float* __restrict__ stats,
                              const float* __restrict__ g, const float* __restrict__ b,
                              int M)
{
    int gid = blockIdx.x * 256 + threadIdx.x;
    if (gid >= M * HDIM) return;
    int c = gid & 127;
    float inv = 1.f / (float)M;
    float mean = stats[c] * inv;
    float var  = stats[128 + c] * inv - mean * mean;
    float v = (x[gid] - mean) * rsqrtf(var + EPSV) * g[c] + b[c];
    y[gid] = v > 0.f ? v : 0.f;
}

// ---------------------------------------------------------------------------
// seq[t,u,d] = users[t,u,d] + sinusoidal PE(t,d)
// ---------------------------------------------------------------------------
__global__ void add_posenc(const float* __restrict__ users, float* __restrict__ seq, int total)
{
    int gid = blockIdx.x * 256 + threadIdx.x;
    if (gid >= total) return;
    int d = gid & 127;
    int t = gid / (NUSR * DDIM);
    int i2 = d >> 1;
    float div = expf(-(float)(2 * i2) * (logf(10000.f) / (float)DDIM));
    float ang = (float)t * div;
    float pe = (d & 1) ? cosf(ang) : sinf(ang);
    seq[gid] = users[gid] + pe;
}

// ---------------------------------------------------------------------------
// Attention: T=4 tokens, one wave32 per (user, head), lane = head dim.
// ---------------------------------------------------------------------------
__global__ __launch_bounds__(256)
void attention_kernel(const float* __restrict__ qkv, float* __restrict__ o)
{
    int gw = blockIdx.x * 8 + (threadIdx.x >> 5);
    int lane = threadIdx.x & 31;
    if (gw >= NUSR * NHEAD) return;
    int u = gw >> 2;
    int h = gw & 3;
    float q[TT], k[TT], v[TT];
#pragma unroll
    for (int t = 0; t < TT; ++t) {
        size_t base = ((size_t)t * NUSR + u) * (3 * DDIM) + h * DHEAD + lane;
        q[t] = qkv[base];
        k[t] = qkv[base + DDIM];
        v[t] = qkv[base + 2 * DDIM];
    }
    const float scale = 0.17677669529663687f;  // 1/sqrt(32)
    float sc[TT][TT];
#pragma unroll
    for (int t = 0; t < TT; ++t)
#pragma unroll
        for (int s2 = 0; s2 < TT; ++s2) {
            float d = q[t] * k[s2];
#pragma unroll
            for (int off = 16; off > 0; off >>= 1) d += __shfl_xor(d, off, 32);
            sc[t][s2] = d * scale;
        }
#pragma unroll
    for (int t = 0; t < TT; ++t) {
        float mx = sc[t][0];
#pragma unroll
        for (int s2 = 1; s2 < TT; ++s2) mx = fmaxf(mx, sc[t][s2]);
        float sum = 0.f;
#pragma unroll
        for (int s2 = 0; s2 < TT; ++s2) { sc[t][s2] = expf(sc[t][s2] - mx); sum += sc[t][s2]; }
        float inv = 1.f / sum;
        float out = 0.f;
#pragma unroll
        for (int s2 = 0; s2 < TT; ++s2) out += sc[t][s2] * inv * v[s2];
        o[((size_t)t * NUSR + u) * DDIM + h * DHEAD + lane] = out;
    }
}

// ---------------------------------------------------------------------------
// xio = LayerNorm(xio + res) over D=128. One wave per row, 4 elems per lane.
// ---------------------------------------------------------------------------
__global__ __launch_bounds__(256)
void residual_ln(float* __restrict__ xio, const float* __restrict__ res,
                 const float* __restrict__ g, const float* __restrict__ b, int M)
{
    int row = blockIdx.x * 8 + (threadIdx.x >> 5);
    int lane = threadIdx.x & 31;
    if (row >= M) return;
    size_t base = (size_t)row * DDIM + lane * 4;
    float4 x = *(const float4*)(xio + base);
    float4 r = *(const float4*)(res + base);
    x.x += r.x; x.y += r.y; x.z += r.z; x.w += r.w;
    float s = x.x + x.y + x.z + x.w;
    float q = x.x * x.x + x.y * x.y + x.z * x.z + x.w * x.w;
#pragma unroll
    for (int off = 16; off > 0; off >>= 1) {
        s += __shfl_xor(s, off, 32);
        q += __shfl_xor(q, off, 32);
    }
    float mean = s * (1.f / DDIM);
    float var  = q * (1.f / DDIM) - mean * mean;
    float inv = rsqrtf(var + EPSV);
    int d0 = lane * 4;
    float4 y;
    y.x = (x.x - mean) * inv * g[d0 + 0] + b[d0 + 0];
    y.y = (x.y - mean) * inv * g[d0 + 1] + b[d0 + 1];
    y.z = (x.z - mean) * inv * g[d0 + 2] + b[d0 + 2];
    y.w = (x.w - mean) * inv * g[d0 + 3] + b[d0 + 3];
    *(float4*)(xio + base) = y;
}

__global__ void gelu_kernel(float* __restrict__ x, int total)
{
    int gid = blockIdx.x * 256 + threadIdx.x;
    if (gid >= total) return;
    float v = x[gid];
    x[gid] = 0.5f * v * (1.f + erff(v * 0.7071067811865475f));
}

// ---------------------------------------------------------------------------
// Scoring: one wave per (pos/neg, pair). cosine sim + mlp on [u, sim].
// ---------------------------------------------------------------------------
__global__ __launch_bounds__(256)
void score_kernel(const float* __restrict__ u_pred, const float* __restrict__ e_pred,
                  const int* __restrict__ pu, const int* __restrict__ pe,
                  const int* __restrict__ nu, const int* __restrict__ ne,
                  const float* __restrict__ mlp_w, const float* __restrict__ mlp_b,
                  float* __restrict__ out)
{
    int gw = blockIdx.x * 8 + (threadIdx.x >> 5);
    int lane = threadIdx.x & 31;
    if (gw >= 2 * BSZ) return;
    int j = (gw >= BSZ) ? 1 : 0;
    int i = gw - j * BSZ;
    int ui = j ? nu[i] : pu[i];
    int ei = j ? ne[i] : pe[i];
    float4 uv = *(const float4*)(u_pred + (size_t)ui * DDIM + lane * 4);
    float4 ev = *(const float4*)(e_pred + (size_t)ei * DDIM + lane * 4);
    float4 wv = *(const float4*)(mlp_w + lane * 4);
    float dot = uv.x * ev.x + uv.y * ev.y + uv.z * ev.z + uv.w * ev.w;
    float un  = uv.x * uv.x + uv.y * uv.y + uv.z * uv.z + uv.w * uv.w;
    float en  = ev.x * ev.x + ev.y * ev.y + ev.z * ev.z + ev.w * ev.w;
    float uw  = uv.x * wv.x + uv.y * wv.y + uv.z * wv.z + uv.w * wv.w;
#pragma unroll
    for (int off = 16; off > 0; off >>= 1) {
        dot += __shfl_xor(dot, off, 32);
        un  += __shfl_xor(un, off, 32);
        en  += __shfl_xor(en, off, 32);
        uw  += __shfl_xor(uw, off, 32);
    }
    if (lane == 0) {
        float sim = dot / (fmaxf(sqrtf(un), 1e-8f) * fmaxf(sqrtf(en), 1e-8f));
        out[(size_t)j * BSZ + i] = uw + sim * mlp_w[128] + mlp_b[0];
    }
}

// ---------------------------------------------------------------------------
// Host orchestration
// ---------------------------------------------------------------------------
static inline float* bump(char*& p, size_t nfloats)
{
    float* r = (float*)p;
    p += ((nfloats * sizeof(float) + 255) / 256) * 256;
    return r;
}

extern "C" void kernel_launch(void* const* d_in, const int* in_sizes, int n_in,
                              void* d_out, int out_size, void* d_ws, size_t ws_size,
                              hipStream_t stream)
{
    const float* x_user  = (const float*)d_in[0];
    const float* x_event = (const float*)d_in[1];
    const int* edge_user  = (const int*)d_in[2];
    const int* edge_event = (const int*)d_in[3];
    const int* pos_user   = (const int*)d_in[4];
    const int* pos_event  = (const int*)d_in[5];
    const int* neg_user   = (const int*)d_in[6];
    const int* neg_event  = (const int*)d_in[7];
    const float* Wu_proj = (const float*)d_in[8];
    const float* bu_proj = (const float*)d_in[9];
    const float* We_proj = (const float*)d_in[10];
    const float* be_proj = (const float*)d_in[11];
    const float* u_lin_l = (const float*)d_in[12];
    const float* u_lin_r = (const float*)d_in[13];
    const float* u_bias  = (const float*)d_in[14];
    const float* e_lin_l = (const float*)d_in[15];
    const float* e_lin_r = (const float*)d_in[16];
    const float* e_bias  = (const float*)d_in[17];
    const float* u_bn_g = (const float*)d_in[18];
    const float* u_bn_b = (const float*)d_in[19];
    const float* e_bn_g = (const float*)d_in[20];
    const float* e_bn_b = (const float*)d_in[21];
    const float* Wqkv = (const float*)d_in[22];
    const float* bqkv = (const float*)d_in[23];
    const float* Wo   = (const float*)d_in[24];
    const float* bo   = (const float*)d_in[25];
    const float* ln1_g = (const float*)d_in[26];
    const float* ln1_b = (const float*)d_in[27];
    const float* ln2_g = (const float*)d_in[28];
    const float* ln2_b = (const float*)d_in[29];
    const float* W1 = (const float*)d_in[30];
    const float* b1 = (const float*)d_in[31];
    const float* W2 = (const float*)d_in[32];
    const float* b2 = (const float*)d_in[33];
    const float* mlp_w = (const float*)d_in[34];
    const float* mlp_b = (const float*)d_in[35];

    // ---- workspace layout (GEMM A-operand buffers padded to 128 rows) ----
    char* w = (char*)d_ws;
    float* xin_u  = bump(w, (size_t)NUP * FDIM);   // staged x_user snapshot
    float* xin_e  = bump(w, (size_t)NEP * FDIM);   // staged x_event snapshot
    float* xu     = bump(w, (size_t)NUP * HDIM);
    float* xe     = bump(w, (size_t)NEP * HDIM);
    float* agg    = bump(w, (size_t)NUP * HDIM);
    float* cnt    = bump(w, NUSR);
    float* tmp    = bump(w, (size_t)NUP * HDIM);
    float* stats  = bump(w, 2 * HDIM);
    float* users4 = bump(w, (size_t)TT * NUSR * DDIM);
    float* seq    = bump(w, (size_t)MSEQP * DDIM);
    // big region: qkv [MSEQ,384] | obuf [MSEQP,128]; whole = ff1 [MSEQP,512]
    float* big    = bump(w, (size_t)MSEQP * DFF);
    float* qkv    = big;
    float* obuf   = big + (size_t)MSEQ * (3 * DDIM);
    float* proj   = big;   // alias of qkv region (dead after attention)
    float* ff1    = big;   // alias of full region
    float* ff2    = bump(w, (size_t)MSEQ * DDIM);
    // transposed weights [N,K]
    float* WuT   = bump(w, 128 * 64);
    float* WeT   = bump(w, 128 * 64);
    float* ulT   = bump(w, 2 * 128 * 128);
    float* urT   = bump(w, 2 * 128 * 128);
    float* elT   = bump(w, 2 * 128 * 128);
    float* erT   = bump(w, 2 * 128 * 128);
    float* WqkvT = bump(w, 2 * 384 * 128);
    float* WoT   = bump(w, 2 * 128 * 128);
    float* W1T   = bump(w, 2 * 512 * 128);
    float* W2T   = bump(w, 2 * 128 * 512);

    auto tr = [&](const float* Bm, float* Bt, int K, int N) {
        transpose_f32<<<(K * N + 255) / 256, 256, 0, stream>>>(Bm, Bt, K, N);
    };
    tr(Wu_proj, WuT, FDIM, HDIM);
    tr(We_proj, WeT, FDIM, HDIM);
    for (int i = 0; i < 2; ++i) {
        tr(u_lin_l + (size_t)i * HDIM * HDIM, ulT + (size_t)i * HDIM * HDIM, HDIM, HDIM);
        tr(u_lin_r + (size_t)i * HDIM * HDIM, urT + (size_t)i * HDIM * HDIM, HDIM, HDIM);
        tr(e_lin_l + (size_t)i * HDIM * HDIM, elT + (size_t)i * HDIM * HDIM, HDIM, HDIM);
        tr(e_lin_r + (size_t)i * HDIM * HDIM, erT + (size_t)i * HDIM * HDIM, HDIM, HDIM);
    }
    for (int l = 0; l < NLT; ++l) {
        tr(Wqkv + (size_t)l * DDIM * 3 * DDIM, WqkvT + (size_t)l * DDIM * 3 * DDIM, DDIM, 3 * DDIM);
        tr(Wo + (size_t)l * DDIM * DDIM,       WoT + (size_t)l * DDIM * DDIM,       DDIM, DDIM);
        tr(W1 + (size_t)l * DDIM * DFF,        W1T + (size_t)l * DDIM * DFF,        DDIM, DFF);
        tr(W2 + (size_t)l * DFF * DDIM,        W2T + (size_t)l * DFF * DDIM,        DFF, DDIM);
    }

    auto gemm = [&](const float* A, const float* Bt, float* Cm, const float* bias,
                    int M, int N, int K, int accum) {
        dim3 grid((M + 127) / 128, N / 64);
        gemm_f32_wmma<<<grid, 256, 0, stream>>>(A, Bt, Cm, bias, M, N, K, accum);
    };

    // ---- GNN over snapshots ---------------------------------------------
    for (int s = 0; s < S_SNAP; ++s) {
        hipMemcpyAsync(xin_u, x_user + (size_t)s * NUSR * FDIM,
                       (size_t)NUSR * FDIM * sizeof(float), hipMemcpyDeviceToDevice, stream);
        hipMemcpyAsync(xin_e, x_event + (size_t)s * NEVT * FDIM,
                       (size_t)NEVT * FDIM * sizeof(float), hipMemcpyDeviceToDevice, stream);
        gemm(xin_u, WuT, xu, bu_proj, NUSR, HDIM, FDIM, 0);
        gemm(xin_e, WeT, xe, be_proj, NEVT, HDIM, FDIM, 0);
        const int* eu = edge_user  + (size_t)s * NEDGE;
        const int* ee = edge_event + (size_t)s * NEDGE;
        for (int i = 0; i < 2; ++i) {
            // user update: msg = xe[ee] scattered at eu
            hipMemsetAsync(agg, 0, (size_t)NUSR * HDIM * sizeof(float), stream);
            hipMemsetAsync(cnt, 0, NUSR * sizeof(float), stream);
            scatter_add<<<(NEDGE * 32 + 255) / 256, 256, 0, stream>>>(xe, ee, eu, agg, cnt, NEDGE);
            mean_div<<<(NUSR * HDIM + 255) / 256, 256, 0, stream>>>(agg, cnt, NUSR * HDIM);
            gemm(agg, ulT + (size_t)i * HDIM * HDIM, tmp, u_bias + i * HDIM, NUSR, HDIM, HDIM, 0);
            gemm(xu,  urT + (size_t)i * HDIM * HDIM, tmp, nullptr,           NUSR, HDIM, HDIM, 1);
            hipMemsetAsync(stats, 0, 2 * HDIM * sizeof(float), stream);
            bn_stats<<<256, 256, 0, stream>>>(tmp, stats, NUSR);
            bn_apply_relu<<<(NUSR * HDIM + 255) / 256, 256, 0, stream>>>(
                tmp, xu, stats, u_bn_g + i * HDIM, u_bn_b + i * HDIM, NUSR);
            // event update: msg = xu[eu] scattered at ee
            hipMemsetAsync(agg, 0, (size_t)NEVT * HDIM * sizeof(float), stream);
            hipMemsetAsync(cnt, 0, NEVT * sizeof(float), stream);
            scatter_add<<<(NEDGE * 32 + 255) / 256, 256, 0, stream>>>(xu, eu, ee, agg, cnt, NEDGE);
            mean_div<<<(NEVT * HDIM + 255) / 256, 256, 0, stream>>>(agg, cnt, NEVT * HDIM);
            gemm(agg, elT + (size_t)i * HDIM * HDIM, tmp, e_bias + i * HDIM, NEVT, HDIM, HDIM, 0);
            gemm(xe,  erT + (size_t)i * HDIM * HDIM, tmp, nullptr,           NEVT, HDIM, HDIM, 1);
            hipMemsetAsync(stats, 0, 2 * HDIM * sizeof(float), stream);
            bn_stats<<<256, 256, 0, stream>>>(tmp, stats, NEVT);
            bn_apply_relu<<<(NEVT * HDIM + 255) / 256, 256, 0, stream>>>(
                tmp, xe, stats, e_bn_g + i * HDIM, e_bn_b + i * HDIM, NEVT);
        }
        if (s < TT) {
            hipMemcpyAsync(users4 + (size_t)s * NUSR * DDIM, xu,
                           (size_t)NUSR * DDIM * sizeof(float),
                           hipMemcpyDeviceToDevice, stream);
        }
    }
    // xe now holds e_pred (snapshot S-1 events)

    // ---- transformer over user sequence ---------------------------------
    add_posenc<<<(MSEQ * DDIM + 255) / 256, 256, 0, stream>>>(users4, seq, MSEQ * DDIM);
    for (int l = 0; l < NLT; ++l) {
        gemm(seq, WqkvT + (size_t)l * DDIM * 3 * DDIM, qkv, bqkv + l * 3 * DDIM,
             MSEQ, 3 * DDIM, DDIM, 0);
        attention_kernel<<<(NUSR * NHEAD + 7) / 8, 256, 0, stream>>>(qkv, obuf);
        gemm(obuf, WoT + (size_t)l * DDIM * DDIM, proj, bo + l * DDIM, MSEQ, DDIM, DDIM, 0);
        residual_ln<<<(MSEQ + 7) / 8, 256, 0, stream>>>(seq, proj,
                                                        ln1_g + l * DDIM, ln1_b + l * DDIM, MSEQ);
        gemm(seq, W1T + (size_t)l * DDIM * DFF, ff1, b1 + l * DFF, MSEQ, DFF, DDIM, 0);
        gelu_kernel<<<(MSEQ * DFF + 255) / 256, 256, 0, stream>>>(ff1, MSEQ * DFF);
        gemm(ff1, W2T + (size_t)l * DFF * DDIM, ff2, b2 + l * DDIM, MSEQ, DDIM, DFF, 0);
        residual_ln<<<(MSEQ + 7) / 8, 256, 0, stream>>>(seq, ff2,
                                                        ln2_g + l * DDIM, ln2_b + l * DDIM, MSEQ);
    }

    // ---- scoring ---------------------------------------------------------
    const float* u_pred = seq + (size_t)(TT - 1) * NUSR * DDIM;
    score_kernel<<<(2 * BSZ + 7) / 8, 256, 0, stream>>>(
        u_pred, xe, pos_user, pos_event, neg_user, neg_event,
        mlp_w, mlp_b, (float*)d_out);
}
